// InformationInteraction_8126078124708
// MI455X (gfx1250) — matrix operand order repaired and checked by reading, not compiled, not run on hardware
//
#include <hip/hip_runtime.h>
#include <stdint.h>

#define EPSBN 1e-5f

typedef __bf16 v16bf __attribute__((ext_vector_type(16)));
typedef float  v8f   __attribute__((ext_vector_type(8)));

union FragAB { v16bf v; uint32_t u[8]; };

// round-to-nearest-even f32 -> bf16, packed pair (x -> low 16, y -> high 16)
__device__ __forceinline__ uint32_t pack_bf16(float x, float y) {
  uint32_t a = __float_as_uint(x), b = __float_as_uint(y);
  a += 0x7FFFu + ((a >> 16) & 1u);
  b += 0x7FFFu + ((b >> 16) & 1u);
  return (a >> 16) | (b & 0xFFFF0000u);
}

// ---------------------------------------------------------------------------
// CDNA5 async copy: global -> LDS, 16B per lane, tracked by ASYNCcnt.
// LDS operand is the low 32 bits of the generic pointer (flat->LDS aperture
// truncates addr[31:0], ISA 10.2), global address is a 64-bit VGPR pair.
// ---------------------------------------------------------------------------
__device__ __forceinline__ void async_copy_b128(const uint32_t* g, uint32_t* l) {
  asm volatile("global_load_async_to_lds_b128 %0, %1, off"
               :: "v"((uint32_t)(uintptr_t)l), "v"((uint64_t)(uintptr_t)g)
               : "memory");
}
__device__ __forceinline__ void wait_async0() {
  asm volatile("s_wait_asynccnt 0x0" ::: "memory");
}

// ---------------------------------------------------------------------------
// Unified tiled GEMM, C[M,N] = A[M,K]*B[K,N] per batch (blockIdx.z).
// A and B are bf16 packed-pair matrices: A[m][kp] pitch PA, B[n][kp] pitch PB
// (both "rows of K"), so every stage is a contiguous 16B async copy.
// Block tile 128(M) x 128(N), K-step 32, double-buffered LDS; 8 waves as 4x2,
// each wave a 32x64 tile -> 2 A-frags x 4 B-frags ->
// 8 v_wmma_f32_16x16x32_bf16 per K-step.
//   EPI 2: out = relu(y*inv + shift) + acc  (fused BN+ReLU+residual add)
// ---------------------------------------------------------------------------
template <int EPI, int K, int PA, int PB, int LDC, long SA, long SB, long SC>
__global__ __launch_bounds__(256)
void gemm_async_wmma(const uint32_t* __restrict__ A, const uint32_t* __restrict__ B,
                     float* __restrict__ Cout,
                     const float* __restrict__ yIn, const float* __restrict__ inv,
                     const float* __restrict__ shift)
{
  // pitch 20 u32 (80B): keeps 16B alignment for b128 and spreads bank windows
  __shared__ uint32_t la[2][128][20];   // A tile 128 rows x 16 kpairs
  __shared__ uint32_t lb[2][128][20];   // B tile 128 rows x 16 kpairs

  const int tid  = threadIdx.x;
  const int lane = tid & 31;
  const int w    = tid >> 5;
  const int wm   = w >> 1;       // 0..3 : wave M strip (32 rows)
  const int wn   = w & 1;        // 0..1 : wave N strip (64 cols)
  const int hk   = lane >> 4;    // half-wave select (ISA bf16 VGPR layout)
  const int l16  = lane & 15;

  const int bz = blockIdx.z;
  const int m0 = blockIdx.y * 128;
  const int n0 = blockIdx.x * 128;

  // staging: row r of each tile gets 8 kpairs per half-thread (2 x b128)
  const int am = tid >> 1;          // 0..127
  const int ap = (tid & 1) * 8;     // 0 or 8

  const uint32_t* aSrc = A + (long)bz * SA + (long)(m0 + am) * PA + ap;
  const uint32_t* bSrc = B + (long)bz * SB + (long)(n0 + am) * PB + ap;

  int ibuf = 0;
  auto issue = [&]() {
    async_copy_b128(aSrc,     &la[ibuf][am][ap]);
    async_copy_b128(aSrc + 4, &la[ibuf][am][ap + 4]);
    async_copy_b128(bSrc,     &lb[ibuf][am][ap]);
    async_copy_b128(bSrc + 4, &lb[ibuf][am][ap + 4]);
    aSrc += 16; bSrc += 16; ibuf ^= 1;
  };

  v8f acc[2][4] = {};

  issue();                                  // prime buffer 0
  int rbuf = 0;
  for (int kt = 0; kt < K; kt += 32) {
    wait_async0();                          // my 4 copies for rbuf done
    __syncthreads();                        // everyone's copies done
    if (kt + 32 < K) issue();               // DMA next tile into other buffer

    FragAB a[2], bf[4];
    const int ar = wm * 32 + l16;
    #pragma unroll
    for (int i = 0; i < 2; ++i) {           // A: pairs {0-3|4-7},{8-11|12-15}
      #pragma unroll
      for (int v = 0; v < 4; ++v) {
        a[i].u[v]     = la[rbuf][ar + 16 * i][4 * hk + v];
        a[i].u[4 + v] = la[rbuf][ar + 16 * i][8 + 4 * hk + v];
      }
    }
    #pragma unroll
    for (int j = 0; j < 4; ++j) {           // B: pairs {0-3|4-7},{8-11|12-15}
      #pragma unroll
      for (int v = 0; v < 4; ++v) {
        bf[j].u[v]     = lb[rbuf][wn * 64 + 16 * j + l16][4 * hk + v];
        bf[j].u[4 + v] = lb[rbuf][wn * 64 + 16 * j + l16][8 + 4 * hk + v];
      }
    }
    #pragma unroll
    for (int i = 0; i < 2; ++i)
      #pragma unroll
      for (int j = 0; j < 4; ++j)
        acc[i][j] = __builtin_amdgcn_wmma_f32_16x16x32_bf16(
            false, a[i].v, false, bf[j].v, (short)0, acc[i][j], false, false);
    rbuf ^= 1;
  }

  // ---- epilogue: C/D layout = reg r -> row r + 8*hk, lane -> col ----
  const int mb = m0 + wm * 32;
  const int nb = n0 + wn * 64;
  float* Cp = Cout + (long)bz * SC;
  const float* yp = yIn + (long)bz * SC;
  #pragma unroll
  for (int i = 0; i < 2; ++i) {
    #pragma unroll
    for (int r = 0; r < 8; ++r) {
      const int  row = mb + 16 * i + r + 8 * hk;
      const long ib  = (long)row * LDC + nb + l16;
      float iv = 0.f, sh = 0.f;
      if (EPI == 2) { iv = inv[row]; sh = shift[row]; }
      #pragma unroll
      for (int j = 0; j < 4; ++j) {
        float c = acc[i][j][r];
        if (EPI == 2) c += fmaxf(yp[ib + 16 * j] * iv + sh, 0.f);
        Cp[ib + 16 * j] = c;
      }
    }
  }
}

// ---------------------------------------------------------------------------
// Elementwise f32 -> packed bf16 pairs (contiguous): 8 pairs per thread.
// ---------------------------------------------------------------------------
__global__ __launch_bounds__(256)
void pack_pairs(const float* __restrict__ in, uint32_t* __restrict__ out)
{
  long base = ((long)blockIdx.x * 256 + threadIdx.x) * 8;
  #pragma unroll
  for (int j = 0; j < 8; ++j) {
    float2 f = ((const float2*)in)[base + j];
    out[base + j] = pack_bf16(f.x, f.y);
  }
}

// ---------------------------------------------------------------------------
// Transpose x[b][c][n] -> xT[b][n][c] as bf16 pairs along c. 64x64 tiles.
// ---------------------------------------------------------------------------
__global__ __launch_bounds__(256)
void transpose_pack(const float* __restrict__ x, uint32_t* __restrict__ xT)
{
  __shared__ float t[64][65];
  const int tid = threadIdx.x;
  const int b  = blockIdx.z;
  const int c0 = blockIdx.y * 64;
  const int n0 = blockIdx.x * 64;

  const float* src = x + ((long)b * 512 + c0) * 4096 + n0;
  const int nn = tid & 63, cc0 = tid >> 6;
  #pragma unroll
  for (int i = 0; i < 16; ++i) {
    int cc = cc0 + 4 * i;
    t[cc][nn] = src[(long)cc * 4096 + nn];
  }
  __syncthreads();

  uint32_t* dst = xT + ((long)b * 4096 + n0) * 256 + (c0 >> 1);
  const int cp = tid & 31, nb = tid >> 5;
  #pragma unroll
  for (int i = 0; i < 8; ++i) {
    int n = nb + 8 * i;
    dst[(long)n * 256 + cp] = pack_bf16(t[2 * cp][n], t[2 * cp + 1][n]);
  }
}

// ---------------------------------------------------------------------------
// BN batch stats: one block per channel, reduce 32*4096 values.
// ---------------------------------------------------------------------------
__global__ __launch_bounds__(256)
void bn_stats(const float* __restrict__ y, const float* __restrict__ gamma,
              const float* __restrict__ beta, float* __restrict__ inv,
              float* __restrict__ shift)
{
  __shared__ float ssum[256], ssq[256];
  const int c = blockIdx.x;
  const int tid = threadIdx.x;
  float s = 0.f, q = 0.f;
  for (int b = 0; b < 32; ++b) {
    const float* p = y + ((long)b * 512 + c) * 4096;
    for (int i = tid; i < 4096; i += 256) { float v = p[i]; s += v; q += v * v; }
  }
  ssum[tid] = s; ssq[tid] = q;
  __syncthreads();
  for (int off = 128; off > 0; off >>= 1) {
    if (tid < off) { ssum[tid] += ssum[tid + off]; ssq[tid] += ssq[tid + off]; }
    __syncthreads();
  }
  if (tid == 0) {
    const float n = 32.f * 4096.f;
    float mean = ssum[0] / n;
    float var  = ssq[0] / n - mean * mean;
    float iv   = gamma[c] * rsqrtf(var + EPSBN);
    inv[c]   = iv;
    shift[c] = beta[c] - mean * iv;
  }
}

// ---------------------------------------------------------------------------
// Row softmax over G (rows of 512) -> bf16 packed pairs (WMMA A layout).
// One wave32 per row, 16 values per lane, shfl_xor reductions.
// ---------------------------------------------------------------------------
__global__ __launch_bounds__(256)
void softmax_rows(const float* __restrict__ G, uint32_t* __restrict__ att)
{
  const int tid = threadIdx.x, lane = tid & 31, w = tid >> 5;
  const long row = (long)blockIdx.x * 8 + w;
  const float4* g4 = (const float4*)(G + row * 512 + lane * 16);
  float4 v[4];
  #pragma unroll
  for (int i = 0; i < 4; ++i) v[i] = g4[i];

  float m = -3.4e38f;
  #pragma unroll
  for (int i = 0; i < 4; ++i)
    m = fmaxf(m, fmaxf(fmaxf(v[i].x, v[i].y), fmaxf(v[i].z, v[i].w)));
  #pragma unroll
  for (int d = 16; d; d >>= 1) m = fmaxf(m, __shfl_xor(m, d, 32));

  float s = 0.f;
  #pragma unroll
  for (int i = 0; i < 4; ++i) {
    v[i].x = __expf(v[i].x - m); v[i].y = __expf(v[i].y - m);
    v[i].z = __expf(v[i].z - m); v[i].w = __expf(v[i].w - m);
    s += v[i].x + v[i].y + v[i].z + v[i].w;
  }
  #pragma unroll
  for (int d = 16; d; d >>= 1) s += __shfl_xor(s, d, 32);
  float r = 1.f / s;

  uint32_t* o = att + row * 256 + lane * 8;
  #pragma unroll
  for (int i = 0; i < 4; ++i) {
    o[2 * i]     = pack_bf16(v[i].x * r, v[i].y * r);
    o[2 * i + 1] = pack_bf16(v[i].z * r, v[i].w * r);
  }
}

// ---------------------------------------------------------------------------
extern "C" void kernel_launch(void* const* d_in, const int* in_sizes, int n_in,
                              void* d_out, int out_size, void* d_ws, size_t ws_size,
                              hipStream_t stream)
{
  (void)in_sizes; (void)n_in; (void)out_size; (void)ws_size;
  const float* x     = (const float*)d_in[0];   // [32,512,64,64]
  const float* wc    = (const float*)d_in[1];   // [512,512]
  const float* gamma = (const float*)d_in[2];   // [512]
  const float* beta  = (const float*)d_in[3];   // [512]
  float* out = (float*)d_out;                   // [32,512,64,64]

  char* ws = (char*)d_ws;
  float*    y   = (float*)(ws);                        // 268,435,456 B
  float*    G   = (float*)(ws + 268435456UL);          //  33,554,432 B
  uint32_t* att = (uint32_t*)(ws + 301989888UL);       //  16,777,216 B
  uint32_t* xb  = (uint32_t*)(ws + 318767104UL);       // 134,217,728 B  x bf16 [c][n/2]
  uint32_t* xT  = (uint32_t*)(ws + 452984832UL);       // 134,217,728 B  x^T bf16 [n][c/2]
  uint32_t* wb  = (uint32_t*)(ws + 587202560UL);       //     524,288 B  W bf16 [o][c/2]
  float*    inv = (float*)(ws + 587726848UL);
  float*    shf = inv + 512;

  dim3 blk(256);

  // 0) one-time bf16 packing / transpose
  pack_pairs<<<dim3(16384), blk, 0, stream>>>(x, xb);          // 33.5M pairs
  pack_pairs<<<dim3(64),    blk, 0, stream>>>(wc, wb);         // 131K pairs
  transpose_pack<<<dim3(64, 8, 32), blk, 0, stream>>>(x, xT);

  // 1) y[b] = W @ x[b]   (M=512, K=512(c), N=4096)  B rows from xT
  gemm_async_wmma<0, 512, 256, 256, 4096,
                  0L, 4096L * 256L, 512L * 4096L>
      <<<dim3(32, 4, 32), blk, 0, stream>>>(wb, xT, y, y, nullptr, nullptr);

  // 2) BN batch statistics -> inv/shift
  bn_stats<<<dim3(512), blk, 0, stream>>>(y, gamma, beta, inv, shf);

  // 3) G[b] = xr @ xr^T   (M=N=512, K=4096(n))  A,B rows from xb
  gemm_async_wmma<0, 4096, 2048, 2048, 512,
                  512L * 2048L, 512L * 2048L, 512L * 512L>
      <<<dim3(4, 4, 32), blk, 0, stream>>>(xb, xb, G, G, nullptr, nullptr);

  // 4) att[b] = softmax_rows(G[b])  (bf16 packed pairs)
  softmax_rows<<<dim3(32 * 512 / 8), blk, 0, stream>>>(G, att);

  // 5) out[b] = relu(y[b]*inv + shift) + att[b] @ xr[b]
  gemm_async_wmma<2, 512, 256, 256, 4096,
                  512L * 256L, 4096L * 256L, 512L * 4096L>
      <<<dim3(32, 4, 32), blk, 0, stream>>>(att, xT, out, y, inv, shf);
}
